// AtlasFreeBNT_73821897884331
// MI455X (gfx1250) — matrix operand order repaired and checked by reading, not compiled, not run on hardware
//
#include <hip/hip_runtime.h>
#include <math.h>

// ---------------- problem constants ----------------
#define BB      4
#define NROI    1632
#define DIN     1632
#define DMODEL  256
#define SEQ     3375          // 15^3 tokens
#define NHEAD   4
#define DHEAD   64
#define DFF     1024
#define QKVW    768           // 3*DMODEL
#define NKT     211           // ceil(SEQ/16) key tiles
#define NKC     106           // SKP/32 key chunks for P@V
#define SKP     3392          // 106*32, padded key count
#define SROW    3400          // dynamic-LDS row stride for score strip
#define EPSV    1e-5f
#define MROWS_FNN   (BB*NROI)     // 6528
#define MROWS_TOK   (BB*SEQ)      // 13500
#define GROW    36            // GEMM staged tile row stride (32 + 4 pad dwords)
#define KROW    68            // attn K-tile row stride (64 + 4 pad dwords)
#define VROW    17            // attn V-tile row stride (16 + 1 pad dword)

typedef __attribute__((ext_vector_type(16))) __bf16 v16bf;
typedef __attribute__((ext_vector_type(8)))  float  v8f;
typedef __attribute__((ext_vector_type(4)))  unsigned int v4u;
typedef __attribute__((ext_vector_type(8)))  int    v8i;
typedef __attribute__((ext_vector_type(4)))  int    v4i;

#define WMMA_BF16(a,b,c) \
  __builtin_amdgcn_wmma_f32_16x16x32_bf16(false,(a),false,(b),(short)0,(c),false,false)

#define RFL(x) __builtin_amdgcn_readfirstlane((int)(x))

// A-matrix (16x32, 16-bit) element->K map: lane half h, element e (ISA 7.12.2)
__device__ __forceinline__ int kmapA(int e, int h) { return (e < 8) ? (8*h + e) : (8*h + e + 8); }
// B-matrix (32x16, 16-bit) element->K map: same as A (lane half h, element e)
__device__ __forceinline__ int kmapB(int e, int h) { return (e < 8) ? (8*h + e) : (8*h + e + 8); }

__device__ __forceinline__ float gelu_exact(float x) {
  return 0.5f * x * (1.0f + erff(x * 0.7071067811865475f));
}

// ---------------------------------------------------------------------------
// TDM: DMA a (tile_rows x tile_k float) 2D tile with row stride row_stride
// elements into LDS at byte offset lds_off. Rows >= tensor_rows are
// zero-filled by the TDM OOB rule (free M/SEQ tail handling). pad_int/pad_amt
// are the raw D# codes: pad (pad_amt+1) dwords after every 2<<pad_int dwords,
// chosen per tile so column accesses are bank-conflict-free and fragment runs
// stay 16B-aligned. Descriptor layout per CDNA5 ISA sec 8.3/8.4.
// ---------------------------------------------------------------------------
__device__ __forceinline__ void tdm_issue_2d(unsigned lds_off, const float* gsrc,
                                             int tensor_rows, int row_stride,
                                             unsigned tile_k, unsigned tile_rows,
                                             unsigned pad_int, unsigned pad_amt)
{
  unsigned long long ga = (unsigned long long)(uintptr_t)gsrc;
  v4u g0;
  g0[0] = (unsigned)RFL(1u);                                   // count=1, user mode
  g0[1] = (unsigned)RFL(lds_off);                              // lds_addr [63:32]
  g0[2] = (unsigned)RFL((unsigned)(ga & 0xFFFFFFFFu));         // global_addr[31:0]
  g0[3] = (unsigned)RFL(((unsigned)((ga >> 32) & 0x01FFFFFFu)) // global_addr[56:32]
                        | (2u << 30));                         // type=2 (image)
  v8i g1;
  g1[0] = RFL((2u << 16)              // data_size = 4 bytes
            | (1u << 20)              // pad_enable
            | (pad_int << 22)         // pad interval code
            | (pad_amt << 25));       // pad amount code
  g1[1] = RFL(tile_k << 16);                                   // tensor_dim0
  g1[2] = RFL(((unsigned)tensor_rows & 0xFFFFu) << 16);        // tensor_dim1 lo16
  g1[3] = RFL(((unsigned)tensor_rows >> 16) | (tile_k << 16)); // dim1 hi | tile_dim0
  g1[4] = RFL(tile_rows);                                      // tile_dim1, tile_dim2=0
  g1[5] = RFL((unsigned)row_stride);                           // tensor_dim0_stride lo32
  g1[6] = 0;
  g1[7] = 0;
  v4i z4 = {0, 0, 0, 0};
#if __clang_major__ >= 23
  v8i z8 = {0, 0, 0, 0, 0, 0, 0, 0};
  __builtin_amdgcn_tensor_load_to_lds(g0, g1, z4, z4, z8, 0);
#else
  __builtin_amdgcn_tensor_load_to_lds(g0, g1, z4, z4, 0);
#endif
}

// ---------------------------------------------------------------------------
// Generic WMMA GEMM: out[m,n] = epilogue( sum_k A[m,k]*W[n,k] + bias[n] )
// Block = 128 threads (4 waves); block tile 64(M) x 64(N).
// TDM double-buffers 64x32 A and W tiles into LDS (row stride 36 floats:
// conflict-free columns + 16B-aligned 8-float runs -> ds_load_b128); wave 0
// drives the DMA, s_wait_tensorcnt 2 pipelines chunk i+1 behind the WMMAs of
// chunk i. Each wave computes a 16x64 strip (A-fragment reused across 4 WMMAs).
// ---------------------------------------------------------------------------
__global__ __launch_bounds__(128) void wmma_gemm_kernel(
    const float* __restrict__ A, const float* __restrict__ W,
    const float* __restrict__ bias, const float* __restrict__ res,
    float* __restrict__ out, int M, int N, int K, int act_gelu)
{
  __shared__ float lds_a[2][64 * GROW];
  __shared__ float lds_b[2][64 * GROW];

  const int lane = threadIdx.x & 31;
  const int wv   = threadIdx.x >> 5;
  const int lm   = lane & 15;
  const int lh   = lane >> 4;

  const int mblk = blockIdx.x * 64;
  const int n0   = blockIdx.y * 64;
  const int m0   = mblk + wv * 16;

  const float* Ab = A + (size_t)mblk * K;
  const float* Wb = W + (size_t)n0 * K;
  const int arows = M - mblk;          // TDM zero-fills rows >= arows
  const int nk    = K / 32;

  v8f acc[4] = {};

  if (wv == 0) {
    tdm_issue_2d((unsigned)(uintptr_t)&lds_a[0][0], Ab, arows, K, 32, 64, 4, 3);
    tdm_issue_2d((unsigned)(uintptr_t)&lds_b[0][0], Wb, 64,    K, 32, 64, 4, 3);
  }

  for (int i = 0; i < nk; ++i) {
    if (wv == 0) {
      if (i + 1 < nk) {
        const int nb = (i + 1) & 1;
        tdm_issue_2d((unsigned)(uintptr_t)&lds_a[nb][0], Ab + (size_t)(i + 1) * 32, arows, K, 32, 64, 4, 3);
        tdm_issue_2d((unsigned)(uintptr_t)&lds_b[nb][0], Wb + (size_t)(i + 1) * 32, 64,    K, 32, 64, 4, 3);
        __builtin_amdgcn_s_wait_tensorcnt(2);   // oldest stage complete
      } else {
        __builtin_amdgcn_s_wait_tensorcnt(0);
      }
    }
    __syncthreads();

    const int buf = i & 1;
    v16bf af;
#pragma unroll
    for (int e = 0; e < 16; ++e)
      af[e] = (__bf16)lds_a[buf][(wv * 16 + lm) * GROW + kmapA(e, lh)];
#pragma unroll
    for (int t = 0; t < 4; ++t) {
      v16bf bfr;
#pragma unroll
      for (int e = 0; e < 16; ++e)
        bfr[e] = (__bf16)lds_b[buf][(16 * t + lm) * GROW + kmapB(e, lh)];
      acc[t] = WMMA_BF16(af, bfr, acc[t]);
    }
    __syncthreads();   // all waves done with buf before it is overwritten
  }

#pragma unroll
  for (int t = 0; t < 4; ++t) {
    const int n = n0 + 16 * t + lm;
    const float bv = bias ? bias[n] : 0.0f;
#pragma unroll
    for (int r = 0; r < 8; ++r) {
      const int m = m0 + r + 8 * lh;
      if (m < M) {
        float v = acc[t][r] + bv;
        if (act_gelu) v = gelu_exact(v);
        if (res) v += res[(size_t)m * N + n];
        out[(size_t)m * N + n] = v;
      }
    }
  }
}

// ---------------------------------------------------------------------------
// Gather (voxel -> ROI embedding) fused with 3x3x3 sum-pool, stride 2.
// grid = B*15^3 blocks, 256 threads (one per channel).
// ---------------------------------------------------------------------------
__global__ __launch_bounds__(256) void gather_pool_kernel(
    const int* __restrict__ C, const float* __restrict__ Qe, float* __restrict__ tok)
{
  const int bs = blockIdx.x;
  const int b  = bs / SEQ;
  const int s  = bs % SEQ;
  const int ox = s / 225, oy = (s / 15) % 15, oz = s % 15;
  const int d  = threadIdx.x;
  float acc = 0.0f;
#pragma unroll
  for (int dx = 0; dx < 3; ++dx)
#pragma unroll
    for (int dy = 0; dy < 3; ++dy)
#pragma unroll
      for (int dz = 0; dz < 3; ++dz) {
        const int x = 2*ox + dx, y = 2*oy + dy, z = 2*oz + dz;
        const int roi = C[((b*32 + x)*32 + y)*32 + z];
        acc += Qe[((size_t)b*NROI + roi) * DMODEL + d];
      }
  tok[((size_t)b*SEQ + s) * DMODEL + d] = acc;
}

// ---------------------------------------------------------------------------
// Flash-style attention. One block = one (b, head, 16-query tile).
// Block = 128 threads (4 waves).
//  - Dynamic LDS: full 16 x SKP score strip (217.6 KB).
//  - Static LDS: per-wave double-buffered TDM tiles for K (16x64, stride 68)
//    and V (32x16 per-wave column slice, stride 17). TENSORcnt is per-wave,
//    so each wave runs its own DMA pipeline with NO extra barriers; the TDM
//    tensor_dim1 OOB rule zero-fills the key tail (no clamps needed).
// Total LDS = 217.6K + 52.2K = 269.8 KB < 320 KB/WGP.
// ---------------------------------------------------------------------------
__global__ __launch_bounds__(128) void attn_kernel(
    const float* __restrict__ qkv, float* __restrict__ O)
{
  extern __shared__ float sm[];
  __shared__ float lds_k[4][2][16 * KROW];
  __shared__ float lds_v[4][2][32 * VROW];

  const int lane = threadIdx.x & 31;
  const int wv   = threadIdx.x >> 5;
  const int lm   = lane & 15;
  const int lh   = lane >> 4;

  const int q0   = blockIdx.x * 16;
  const int head = blockIdx.y;
  const int b    = blockIdx.z;
  const float* base = qkv + (size_t)b * SEQ * QKVW;

  // ---- Q fragments (reused over all key tiles): two K=32 chunks of DHEAD=64
  const int qrow = min(q0 + lm, SEQ - 1);
  const float* qp = base + (size_t)qrow * QKVW + head * DHEAD;
  v16bf qa0, qa1;
#pragma unroll
  for (int e = 0; e < 16; ++e) {
    const int d = kmapA(e, lh);
    qa0[e] = (__bf16)qp[d];
    qa1[e] = (__bf16)qp[d + 32];
  }

  // ---- Phase 1: scores -> LDS (per-wave TDM pipeline for K tiles)
  const float* kg = base + DMODEL + head * DHEAD;   // K matrix base
  tdm_issue_2d((unsigned)(uintptr_t)&lds_k[wv][0][0], kg + (size_t)(wv*16) * QKVW,
               SEQ - wv*16, QKVW, 64, 16, 5, 3);
  int it = 0;
  for (int kt = wv; kt < NKT; kt += 4, ++it) {
    const int ktn = kt + 4;
    if (ktn < NKT) {
      tdm_issue_2d((unsigned)(uintptr_t)&lds_k[wv][(it+1)&1][0], kg + (size_t)(ktn*16) * QKVW,
                   SEQ - ktn*16, QKVW, 64, 16, 5, 3);
      __builtin_amdgcn_s_wait_tensorcnt(1);
    } else {
      __builtin_amdgcn_s_wait_tensorcnt(0);
    }
    const float* kl = &lds_k[wv][it & 1][0];
    v16bf kb0, kb1;
#pragma unroll
    for (int e = 0; e < 16; ++e) {
      const int d = kmapB(e, lh);
      kb0[e] = (__bf16)kl[lm * KROW + d];
      kb1[e] = (__bf16)kl[lm * KROW + 32 + d];
    }
    v8f c = {};
    c = WMMA_BF16(qa0, kb0, c);
    c = WMMA_BF16(qa1, kb1, c);
#pragma unroll
    for (int r = 0; r < 8; ++r)
      sm[(r + 8*lh) * SROW + kt*16 + lm] = c[r] * 0.125f; // 1/sqrt(64)
  }
  __syncthreads();

  // ---- Phase 2: softmax (wave wv handles rows 4wv..4wv+3)
  for (int rr = 0; rr < 4; ++rr) {
    float* rowp = sm + (4*wv + rr) * SROW;
    float mx = -INFINITY;
    for (int kk = lane; kk < SEQ; kk += 32) mx = fmaxf(mx, rowp[kk]);
#pragma unroll
    for (int off = 16; off > 0; off >>= 1) mx = fmaxf(mx, __shfl_xor(mx, off, 32));
    float ssum = 0.0f;
    for (int kk = lane; kk < SEQ; kk += 32) {
      const float p = expf(rowp[kk] - mx);
      rowp[kk] = p;
      ssum += p;
    }
#pragma unroll
    for (int off = 16; off > 0; off >>= 1) ssum += __shfl_xor(ssum, off, 32);
    const float inv = 1.0f / ssum;
    for (int kk = lane; kk < SEQ; kk += 32) rowp[kk] *= inv;
    for (int kk = SEQ + lane; kk < SKP; kk += 32) rowp[kk] = 0.0f; // pad -> P=0
  }
  __syncthreads();

  // ---- Phase 3: O[:, 16wv..16wv+16) = P @ V (per-wave TDM pipeline for V)
  const float* vg = base + 2*DMODEL + head*DHEAD + 16*wv;   // this wave's V cols
  tdm_issue_2d((unsigned)(uintptr_t)&lds_v[wv][0][0], vg, SEQ, QKVW, 16, 32, 3, 0);
  v8f o = {};
  for (int kc = 0; kc < NKC; ++kc) {
    if (kc + 1 < NKC) {
      tdm_issue_2d((unsigned)(uintptr_t)&lds_v[wv][(kc+1)&1][0], vg + (size_t)(kc+1)*32*QKVW,
                   SEQ - (kc+1)*32, QKVW, 16, 32, 3, 0);
      __builtin_amdgcn_s_wait_tensorcnt(1);
    } else {
      __builtin_amdgcn_s_wait_tensorcnt(0);
    }
    const float* vl = &lds_v[wv][kc & 1][0];
    v16bf pa, vb;
#pragma unroll
    for (int e = 0; e < 16; ++e) {
      pa[e] = (__bf16)sm[lm * SROW + kc*32 + kmapA(e, lh)];
      vb[e] = (__bf16)vl[kmapB(e, lh) * VROW + lm];
    }
    o = WMMA_BF16(pa, vb, o);
  }
  const int dc = head*DHEAD + 16*wv + lm;
#pragma unroll
  for (int r = 0; r < 8; ++r) {
    const int q = q0 + r + 8*lh;
    if (q < SEQ)
      O[((size_t)b*SEQ + q) * DMODEL + dc] = o[r];
  }
}

// ---------------------------------------------------------------------------
// In-place LayerNorm over rows of 256. grid = rows, block = 256.
// ---------------------------------------------------------------------------
__global__ __launch_bounds__(256) void ln_kernel(
    float* __restrict__ x, const float* __restrict__ sc, const float* __restrict__ bi)
{
  __shared__ float red[256];
  const int tid = threadIdx.x;
  float* rowp = x + (size_t)blockIdx.x * DMODEL;
  const float v = rowp[tid];

  red[tid] = v; __syncthreads();
  for (int st = 128; st > 0; st >>= 1) { if (tid < st) red[tid] += red[tid + st]; __syncthreads(); }
  const float mean = red[0] / (float)DMODEL;
  __syncthreads();

  const float dvv = v - mean;
  red[tid] = dvv * dvv; __syncthreads();
  for (int st = 128; st > 0; st >>= 1) { if (tid < st) red[tid] += red[tid + st]; __syncthreads(); }
  const float var = red[0] / (float)DMODEL;

  rowp[tid] = dvv * rsqrtf(var + EPSV) * sc[tid] + bi[tid];
}

// mean over sequence: grid = B, block = 256
__global__ __launch_bounds__(256) void mean_kernel(
    const float* __restrict__ x, float* __restrict__ xm)
{
  const int b = blockIdx.x, d = threadIdx.x;
  float acc = 0.0f;
  for (int s = 0; s < SEQ; ++s) acc += x[((size_t)b*SEQ + s) * DMODEL + d];
  xm[b * DMODEL + d] = acc * (1.0f / (float)SEQ);
}

// classifier: 8 outputs total
__global__ void logits_kernel(const float* __restrict__ xm, const float* __restrict__ Wc,
                              const float* __restrict__ bc, float* __restrict__ out)
{
  const int i = threadIdx.x;
  if (i < BB * 2) {
    const int b = i >> 1, c = i & 1;
    float acc = bc[c];
    for (int d = 0; d < DMODEL; ++d) acc += xm[b * DMODEL + d] * Wc[c * DMODEL + d];
    out[i] = acc;
  }
}

// ---------------------------------------------------------------------------
extern "C" void kernel_launch(void* const* d_in, const int* in_sizes, int n_in,
                              void* d_out, int out_size, void* d_ws, size_t ws_size,
                              hipStream_t stream) {
  (void)in_sizes; (void)n_in; (void)out_size; (void)ws_size;
  const int*   C         = (const int*)  d_in[0];
  const float* F         = (const float*)d_in[1];
  const float* W_fnn     = (const float*)d_in[2];
  const float* b_fnn     = (const float*)d_in[3];
  const float* in_proj_w = (const float*)d_in[4];
  const float* in_proj_b = (const float*)d_in[5];
  const float* out_w     = (const float*)d_in[6];
  const float* out_b     = (const float*)d_in[7];
  const float* ln1_s     = (const float*)d_in[8];
  const float* ln1_b     = (const float*)d_in[9];
  const float* W1        = (const float*)d_in[10];
  const float* b1        = (const float*)d_in[11];
  const float* W2        = (const float*)d_in[12];
  const float* b2        = (const float*)d_in[13];
  const float* ln2_s     = (const float*)d_in[14];
  const float* ln2_b     = (const float*)d_in[15];
  const float* on_s      = (const float*)d_in[16];
  const float* on_b      = (const float*)d_in[17];
  const float* Wc        = (const float*)d_in[18];
  const float* bc        = (const float*)d_in[19];
  float* out = (float*)d_out;

  // workspace layout (floats)
  float* w    = (float*)d_ws;
  float* Qe   = w;                                   // 6528  x 256
  float* tok  = Qe   + (size_t)MROWS_FNN * DMODEL;   // 13500 x 256
  float* qkv  = tok  + (size_t)MROWS_TOK * DMODEL;   // 13500 x 768
  float* attn = qkv  + (size_t)MROWS_TOK * QKVW;     // 13500 x 256
  float* xb   = attn + (size_t)MROWS_TOK * DMODEL;   // 13500 x 256
  float* hb   = xb   + (size_t)MROWS_TOK * DMODEL;   // 13500 x 1024
  float* yb   = hb   + (size_t)MROWS_TOK * DFF;      // 13500 x 256
  float* xm   = yb   + (size_t)MROWS_TOK * DMODEL;   // 4 x 256

  // 1) ROI FNN: Qe = gelu(F @ W_fnn^T + b_fnn)
  wmma_gemm_kernel<<<dim3((MROWS_FNN + 63)/64, DMODEL/64), 128, 0, stream>>>(
      F, W_fnn, b_fnn, nullptr, Qe, MROWS_FNN, DMODEL, DIN, 1);

  // 2) gather + 3^3 sum-pool -> tokens
  gather_pool_kernel<<<BB * SEQ, 256, 0, stream>>>(C, Qe, tok);

  // 3) QKV projection
  wmma_gemm_kernel<<<dim3((MROWS_TOK + 63)/64, QKVW/64), 128, 0, stream>>>(
      tok, in_proj_w, in_proj_b, nullptr, qkv, MROWS_TOK, QKVW, DMODEL, 0);

  // 4) attention (217.6 KB dynamic LDS per block + static TDM tiles)
  attn_kernel<<<dim3(NKT, NHEAD, BB), 128, (size_t)16 * SROW * sizeof(float), stream>>>(qkv, attn);

  // 5) out-proj + residual(tokens), then LN1 in place -> xb
  wmma_gemm_kernel<<<dim3((MROWS_TOK + 63)/64, DMODEL/64), 128, 0, stream>>>(
      attn, out_w, out_b, tok, xb, MROWS_TOK, DMODEL, DMODEL, 0);
  ln_kernel<<<MROWS_TOK, 256, 0, stream>>>(xb, ln1_s, ln1_b);

  // 6) FFN
  wmma_gemm_kernel<<<dim3((MROWS_TOK + 63)/64, DFF/64), 128, 0, stream>>>(
      xb, W1, b1, nullptr, hb, MROWS_TOK, DFF, DMODEL, 1);
  wmma_gemm_kernel<<<dim3((MROWS_TOK + 63)/64, DMODEL/64), 128, 0, stream>>>(
      hb, W2, b2, xb, yb, MROWS_TOK, DMODEL, DFF, 0);
  ln_kernel<<<MROWS_TOK, 256, 0, stream>>>(yb, ln2_s, ln2_b);

  // 7) output norm, mean over S, classifier
  ln_kernel<<<MROWS_TOK, 256, 0, stream>>>(yb, on_s, on_b);
  mean_kernel<<<BB, 256, 0, stream>>>(yb, xm);
  logits_kernel<<<1, 32, 0, stream>>>(xm, Wc, bc, out);
}
